// PreCoBlock_8358006358687
// MI455X (gfx1250) — compile-verified, matrix operand order checked by reading, not run on hardware
//
#include <hip/hip_runtime.h>
#include <hip/hip_bf16.h>
#include <math.h>

// ---------------------------------------------------------------------------
// Shapes (compile-time constants from the reference)
// ---------------------------------------------------------------------------
#define BB 4
#define LL 2048
#define DM 512
#define DI 3072            // D_INNER
#define DT_RANK 32
#define D_STATE 8
#define NHEAD 8
#define HF 64
#define MROWS (BB*LL)      // 8192

typedef __attribute__((ext_vector_type(16))) __bf16 v16bf;
typedef __attribute__((ext_vector_type(8)))  float  v8f;

// ---------------------------------------------------------------------------
// Helpers
// ---------------------------------------------------------------------------
__device__ __forceinline__ unsigned short f2bf(float f) {
  unsigned int u = __builtin_bit_cast(unsigned int, f);
  u += 0x7FFFu + ((u >> 16) & 1u);           // round-to-nearest-even
  return (unsigned short)(u >> 16);
}
__device__ __forceinline__ unsigned int pack2bf(float a, float b) {
  return (unsigned int)f2bf(a) | ((unsigned int)f2bf(b) << 16);
}
__device__ __forceinline__ float sigmf(float x) { return 1.0f / (1.0f + __expf(-x)); }
__device__ __forceinline__ float siluf(float x) { return x * sigmf(x); }

// CDNA5 async copy: global -> LDS, 16 bytes per enabled lane (ASYNCcnt).
__device__ __forceinline__ void async_b128(unsigned lds_byte_off, const void* gptr) {
  asm volatile("global_load_async_to_lds_b128 %0, %1, off"
               :: "v"(lds_byte_off), "v"((unsigned long long)(size_t)gptr)
               : "memory");
}
__device__ __forceinline__ void wait_async0() {
#if __has_builtin(__builtin_amdgcn_s_wait_asynccnt)
  __builtin_amdgcn_s_wait_asynccnt(0);
#else
  asm volatile("s_wait_asynccnt 0" ::: "memory");
#endif
}
// Generic (flat) LDS pointer -> 32-bit LDS byte offset (low 32 bits of aperture addr).
__device__ __forceinline__ unsigned lds_off(const void* p) {
  return (unsigned)(unsigned long long)(size_t)p;
}

__device__ __forceinline__ float warp_sum(float v) {
#pragma unroll
  for (int o = 16; o > 0; o >>= 1) v += __shfl_xor(v, o, 32);
  return v;
}
__device__ __forceinline__ float block_sum256(float v, float* sh) {
  v = warp_sum(v);
  int lane = threadIdx.x & 31, w = threadIdx.x >> 5;
  if (lane == 0) sh[w] = v;
  __syncthreads();
  float r = (threadIdx.x < 8) ? sh[threadIdx.x] : 0.0f;
  if (w == 0) { r = warp_sum(r); if (lane == 0) sh[0] = r; }
  __syncthreads();
  float out = sh[0];
  __syncthreads();
  return out;
}

// ---------------------------------------------------------------------------
// Kernel: streaming fp32 -> bf16 conversion (one pass per GEMM operand)
// ---------------------------------------------------------------------------
__global__ __launch_bounds__(256) void f32_to_bf16_kernel(
    const float* __restrict__ src, unsigned short* __restrict__ dst, long long n4) {
  long long i = (long long)blockIdx.x * 256 + threadIdx.x;
  if (i >= n4) return;
  float4 f = ((const float4*)src)[i];
  uint2 o;
  o.x = pack2bf(f.x, f.y);
  o.y = pack2bf(f.z, f.w);
  ((uint2*)dst)[i] = o;
}

// ---------------------------------------------------------------------------
// Kernel: RMSNorm  xn = x * rsqrt(mean(x^2)+1e-5) * w
// ---------------------------------------------------------------------------
__global__ __launch_bounds__(256) void rmsnorm_kernel(
    const float* __restrict__ x, const float* __restrict__ w, float* __restrict__ xn) {
  __shared__ float sh[8];
  size_t row = blockIdx.x;
  const float* xr = x + row * DM;
  float v0 = xr[threadIdx.x];
  float v1 = xr[threadIdx.x + 256];
  float ss = block_sum256(v0 * v0 + v1 * v1, sh);
  float scale = rsqrtf(ss * (1.0f / DM) + 1e-5f);
  float* o = xn + row * DM;
  o[threadIdx.x]       = v0 * scale * w[threadIdx.x];
  o[threadIdx.x + 256] = v1 * scale * w[threadIdx.x + 256];
}

// ---------------------------------------------------------------------------
// WMMA GEMM, bf16 operands: C[M,N] = A[M,K] * B[N,K]^T (+bias)
// Tiles staged into LDS with global_load_async_to_lds_b128, double-buffered:
// async engine fetches tile k+1 while WMMAs consume tile k.
// Block 256 = 8 waves (2x4), tile 64(M) x 128(N), K-step 32.
// ---------------------------------------------------------------------------
#define GBM 64
#define GBN 128
#define GBK 32
#define LDSW 40   // padded LDS row stride in bf16 elems (80B: 16B-aligned chunks, no bank conflicts)

__device__ __forceinline__ void issue_tiles(
    const unsigned short* __restrict__ A, int lda,
    const unsigned short* __restrict__ Bw, int ldb,
    int block_m, int block_n, int N, int k0,
    unsigned sA_off, unsigned sB_off, int wave, int lane) {
  int r = wave * 8 + (lane >> 2);   // 8 rows per wave-instruction
  int c = lane & 3;                 // 4 x 16B chunks per 64B row
  // A tile: 64 x 32 bf16, always in-range
  async_b128(sA_off + (unsigned)(r * (LDSW * 2) + c * 16),
             A + (size_t)(block_m + r) * lda + k0 + c * 8);
  // B tile: 128 x 32 bf16, bounds-checked on N (disabled lanes transfer nothing)
#pragma unroll
  for (int j = 0; j < 2; ++j) {
    int rb = wave * 16 + j * 8 + (lane >> 2);
    if (block_n + rb < N)
      async_b128(sB_off + (unsigned)(rb * (LDSW * 2) + c * 16),
                 Bw + (size_t)(block_n + rb) * ldb + k0 + c * 8);
  }
}

__global__ __launch_bounds__(256) void gemm_bf16_wmma(
    const unsigned short* __restrict__ A, int lda,
    const unsigned short* __restrict__ Bw, int ldb,
    float* __restrict__ C, int ldc,
    int N, int K, const float* __restrict__ bias) {
  __shared__ unsigned short sA[2][GBM * LDSW];
  __shared__ unsigned short sB[2][GBN * LDSW];
  const int tid  = threadIdx.x;
  const int wave = tid >> 5, lane = tid & 31;
  const int wm = wave >> 2, wn = wave & 3;
  const int block_m = blockIdx.y * GBM;
  const int block_n = blockIdx.x * GBN;
  const int lrow = lane & 15;
  const int khi  = (lane >> 4) * 8;
  const int hi8  = (lane >> 4) * 8;
  unsigned offA[2] = { lds_off(&sA[0][0]), lds_off(&sA[1][0]) };
  unsigned offB[2] = { lds_off(&sB[0][0]), lds_off(&sB[1][0]) };

  v8f acc[2][2] = {};
  const int nk = K / GBK;

  issue_tiles(A, lda, Bw, ldb, block_m, block_n, N, 0, offA[0], offB[0], wave, lane);
  wait_async0();
  __syncthreads();

  for (int kt = 0; kt < nk; ++kt) {
    const int cur = kt & 1;
    if (kt + 1 < nk)
      issue_tiles(A, lda, Bw, ldb, block_m, block_n, N, (kt + 1) * GBK,
                  offA[cur ^ 1], offB[cur ^ 1], wave, lane);

    union Frag { uint4 u[2]; v16bf v; };
    Frag af[2], bf[2];
#pragma unroll
    for (int t = 0; t < 2; ++t) {
      const unsigned short* pa = &sA[cur][(wm * 32 + t * 16 + lrow) * LDSW + khi];
      af[t].u[0] = *(const uint4*)(pa);
      af[t].u[1] = *(const uint4*)(pa + 16);
      const unsigned short* pb = &sB[cur][(wn * 32 + t * 16 + lrow) * LDSW + khi];
      bf[t].u[0] = *(const uint4*)(pb);
      bf[t].u[1] = *(const uint4*)(pb + 16);
    }
#pragma unroll
    for (int i = 0; i < 2; ++i)
#pragma unroll
      for (int j = 0; j < 2; ++j)
        acc[i][j] = __builtin_amdgcn_wmma_f32_16x16x32_bf16(
            false, af[i].v, false, bf[j].v, (short)0, acc[i][j], false, false);

    if (kt + 1 < nk) wait_async0();
    __syncthreads();
  }

#pragma unroll
  for (int i = 0; i < 2; ++i)
#pragma unroll
    for (int j = 0; j < 2; ++j) {
      int n = block_n + wn * 32 + j * 16 + lrow;
      if (n < N) {
        float bv = bias ? bias[n] : 0.0f;
#pragma unroll
        for (int e = 0; e < 8; ++e) {
          int m = block_m + wm * 32 + i * 16 + hi8 + e;
          C[(size_t)m * ldc + n] = acc[i][j][e] + bv;
        }
      }
    }
}

// ---------------------------------------------------------------------------
// Kernel: causal depthwise conv (width 3) + bias + SiLU
// ---------------------------------------------------------------------------
__global__ __launch_bounds__(256) void conv_silu_kernel(
    const float* __restrict__ xz, const float* __restrict__ cw,
    const float* __restrict__ cb, float* __restrict__ xb) {
  size_t idx = (size_t)blockIdx.x * 256 + threadIdx.x;
  if (idx >= (size_t)MROWS * DI) return;
  int c = (int)(idx % DI);
  size_t row = idx / DI;
  int l = (int)(row % LL);
  float acc = cb[c];
#pragma unroll
  for (int j = 0; j < 3; ++j) {
    int dlt = j - 2;
    if (l + dlt >= 0) acc += cw[c * 3 + j] * xz[(row + dlt) * (2 * DI) + c];
  }
  xb[idx] = siluf(acc);
}

// ---------------------------------------------------------------------------
// Kernel: Longhorn scan. One thread per (b,d); k/q for step l+1 async-copied
// into an LDS double buffer while step l computes.
// ---------------------------------------------------------------------------
__global__ __launch_bounds__(256) void longhorn_scan_kernel(
    float* __restrict__ xb, const float* __restrict__ xz,
    const float* __restrict__ dt, const float* __restrict__ xdbl,
    const float* __restrict__ dtb, const float* __restrict__ Dp) {
  int d = blockIdx.x * 256 + threadIdx.x;
  int b = blockIdx.y;
  __shared__ float skq2[2][16];
  float s[D_STATE];
#pragma unroll
  for (int n = 0; n < D_STATE; ++n) s[n] = 0.0f;
  const float dval = Dp[d];
  const float dbias = dtb[d];
  // prologue: step 0 k/q
  if (threadIdx.x < 16) skq2[0][threadIdx.x] = xdbl[(size_t)b * LL * 48 + 32 + threadIdx.x];
  __syncthreads();
  for (int l = 0; l < LL; ++l) {
    size_t row = (size_t)b * LL + l;
    const float* skq = skq2[l & 1];
    if (l + 1 < LL && threadIdx.x < 4)
      async_b128(lds_off(&skq2[(l + 1) & 1][0]) + (unsigned)threadIdx.x * 16,
                 xdbl + (row + 1) * 48 + 32 + threadIdx.x * 4);
    float kk = 0.0f;
#pragma unroll
    for (int n = 0; n < D_STATE; ++n) kk += skq[n] * skq[n];
    float x  = xb[row * DI + d];
    float z  = xz[row * (2 * DI) + DI + d];
    float sg = sigmf(dt[row * DI + d] + dbias);
    sg = sg / (1.0f + sg * kk);
    float yv = 0.0f;
#pragma unroll
    for (int n = 0; n < D_STATE; ++n) {
      float kn = skq[n];
      float dBn = sg * kn;
      s[n] = s[n] * (1.0f - dBn * kn) + x * dBn;
      yv += s[n] * skq[8 + n];
    }
    yv += dval * x;
    xb[row * DI + d] = yv * siluf(z);
    if (l + 1 < LL) {
      __builtin_prefetch(xb + (row + 1) * DI + d, 0, 1);
      __builtin_prefetch(dt + (row + 1) * DI + d, 0, 1);
      __builtin_prefetch(xz + (row + 1) * (2 * DI) + DI + d, 0, 1);
      wait_async0();
    }
    __syncthreads();
  }
}

// ---------------------------------------------------------------------------
// Kernel: TTT scan. 32 blocks (b*NH+h), 64 threads; thread g owns W[:,g] in
// 64 VGPRs. xk/xv rows for t+1 async-copied to LDS while step t computes.
// Z1q = Z1 - eta*grad*(||xk||^2 + 1) removes the second matvec.
// ---------------------------------------------------------------------------
__global__ __launch_bounds__(64) void ttt_scan_kernel(
    const float* __restrict__ proj, const float* __restrict__ lr_bias,
    const float* __restrict__ W1, const float* __restrict__ b1,
    const float* __restrict__ token_idx, const float* __restrict__ token_bias,
    const float* __restrict__ lnw_all, const float* __restrict__ lnb_all,
    float* __restrict__ outb) {
  const int bh = blockIdx.x, b = bh >> 3, h = bh & 7;
  const int g = threadIdx.x;
  float W[HF];
#pragma unroll
  for (int f = 0; f < HF; ++f) W[f] = W1[(h * HF + f) * HF + g];
  float bb  = b1[h * HF + g];
  const float lnw = lnw_all[h * HF + g];
  const float lnb = lnb_all[h * HF + g];
  const float lrb = lr_bias[h];
  __shared__ float sxk2[2][HF], sxv2[2][HF], sZ[HF], sG[HF], sGX[HF];
  const float eps = 1e-6f;
  {
    size_t ro0 = (size_t)b * LL * 1544;
    sxk2[0][g] = proj[ro0 + h * HF + g];
    sxv2[0][g] = proj[ro0 + DM + h * HF + g];
  }
  __syncthreads();

  for (int t = 0; t < LL; ++t) {
    const int cur = t & 1;
    const float* sxk = sxk2[cur];
    const float* sxv = sxv2[cur];
    size_t ro = ((size_t)b * LL + t) * 1544;
    if (t + 1 < LL && threadIdx.x < 32) {
      size_t ron = ro + 1544;
      int ln = threadIdx.x;
      const float* gp = (ln < 16) ? (proj + ron + h * HF + ln * 4)
                                  : (proj + ron + DM + h * HF + (ln - 16) * 4);
      unsigned lo = (ln < 16) ? lds_off(&sxk2[cur ^ 1][0]) + (unsigned)ln * 16
                              : lds_off(&sxv2[cur ^ 1][0]) + (unsigned)(ln - 16) * 16;
      async_b128(lo, gp);
    }
    float xk = sxk[g];
    float z1 = bb, kk = 0.0f;
#pragma unroll
    for (int f = 0; f < HF; ++f) { z1 += sxk[f] * W[f]; kk += sxk[f] * sxk[f]; }
    sZ[g] = z1;
    __syncthreads();
    float s1 = 0.0f, s2 = 0.0f;
#pragma unroll
    for (int f = 0; f < HF; ++f) { float v = sZ[f]; s1 += v; s2 += v * v; }
    float mu  = s1 * (1.0f / HF);
    float var = s2 * (1.0f / HF) - mu * mu;
    float rstd = rsqrtf(var + eps);
    float xh = (z1 - mu) * rstd;
    float go = lnw * xh + lnb - (sxv[g] - xk);
    float gxh = go * lnw;
    __syncthreads();
    sG[g] = gxh; sGX[g] = gxh * xh;
    __syncthreads();
    float t1 = 0.0f, t2 = 0.0f;
#pragma unroll
    for (int f = 0; f < HF; ++f) { t1 += sG[f]; t2 += sGX[f]; }
    float grad = ((float)HF * gxh - t1 - xh * t2) * rstd * (1.0f / HF);
    float lrr  = proj[ro + 3 * DM + h];
    float tok  = token_idx[t] + token_bias[t];
    tok = tok > 0.0f ? tok : 0.0f;
    float eg = sigmf(lrr + lrb) * tok * grad;
#pragma unroll
    for (int f = 0; f < HF; ++f) W[f] -= eg * sxk[f];
    bb -= eg;
    float z1q = z1 - eg * (kk + 1.0f);
    __syncthreads();
    sZ[g] = z1q;
    __syncthreads();
    float u1 = 0.0f, u2 = 0.0f;
#pragma unroll
    for (int f = 0; f < HF; ++f) { float v = sZ[f]; u1 += v; u2 += v * v; }
    float mu2  = u1 * (1.0f / HF);
    float var2 = u2 * (1.0f / HF) - mu2 * mu2;
    float outv = xk + lnw * (z1q - mu2) * rsqrtf(var2 + eps) + lnb;
    outb[((size_t)b * LL + t) * DM + h * HF + g] = outv;
    wait_async0();
    __syncthreads();
  }
}

// ---------------------------------------------------------------------------
// Kernel: post-norm + GELU gate
// ---------------------------------------------------------------------------
__global__ __launch_bounds__(256) void gate_kernel(
    const float* __restrict__ ttt_raw, const float* __restrict__ proj,
    const float* __restrict__ pnw, const float* __restrict__ pnb,
    float* __restrict__ gated) {
  __shared__ float sh[8];
  size_t row = blockIdx.x;
  const float* tr = ttt_raw + row * DM;
  float v0 = tr[threadIdx.x], v1 = tr[threadIdx.x + 256];
  float ssum = block_sum256(v0 + v1, sh);
  float ssq  = block_sum256(v0 * v0 + v1 * v1, sh);
  float mu  = ssum * (1.0f / DM);
  float var = ssq * (1.0f / DM) - mu * mu;
  float rstd = rsqrtf(var + 1e-5f);
  const float* gp = proj + row * 1544 + 2 * DM;
  float* o = gated + row * DM;
#pragma unroll
  for (int p = 0; p < 2; ++p) {
    int c = threadIdx.x + p * 256;
    float v = p ? v1 : v0;
    float normed = (v - mu) * rstd * pnw[c] + pnb[c];
    float xg = gp[c];
    float gel = 0.5f * xg * (1.0f + tanhf(0.7978845608f * (xg + 0.044715f * xg * xg * xg)));
    o[c] = gel * normed;
  }
}

// ---------------------------------------------------------------------------
// Kernel: gq_mean[b,d] = mean_l gq[b,l,d]
// ---------------------------------------------------------------------------
__global__ __launch_bounds__(512) void colmean_kernel(
    const float* __restrict__ gq, float* __restrict__ gqm) {
  int b = blockIdx.x, d = threadIdx.x;
  float s = 0.0f;
  const float* p = gq + (size_t)b * LL * DM + d;
  for (int l = 0; l < LL; ++l) s += p[(size_t)l * DM];
  gqm[b * DM + d] = s * (1.0f / LL);
}

// ---------------------------------------------------------------------------
// Kernel: gain + mix + emit (out | residual | gain)
// ---------------------------------------------------------------------------
__global__ __launch_bounds__(256) void out_kernel(
    const float* __restrict__ zl, const float* __restrict__ zt,
    const float* __restrict__ gk, const float* __restrict__ gqm,
    const float* __restrict__ x, const float* __restrict__ alpha,
    const float* __restrict__ beta, float* __restrict__ d_out) {
  __shared__ float sh[8];
  size_t row = blockIdx.x;
  int b = (int)(row >> 11);
  float part = 0.0f;
#pragma unroll
  for (int p = 0; p < 2; ++p) {
    int c = threadIdx.x + p * 256;
    part += gqm[b * DM + c] * gk[row * DM + c];
  }
  float score = block_sum256(part, sh) * (1.0f / 22.62741699796952f);
  float gv = sigmf(alpha[0] * score + beta[0]);
  gv = gv < 0.55f ? gv : 0.55f;
#pragma unroll
  for (int p = 0; p < 2; ++p) {
    int c = threadIdx.x + p * 256;
    float a = zl[row * DM + c];
    float t = zt[row * DM + c];
    d_out[row * DM + c] = a + gv * (t - a);
    d_out[(size_t)MROWS * DM + row * DM + c] = x[row * DM + c];
  }
  if (threadIdx.x == 0) d_out[(size_t)2 * MROWS * DM + row] = gv;
}

// ---------------------------------------------------------------------------
// Host launcher
// ---------------------------------------------------------------------------
static inline void cvt_bf16(const float* s, unsigned short* d, size_t n, hipStream_t st) {
  size_t n4 = n / 4;
  f32_to_bf16_kernel<<<(unsigned)((n4 + 255) / 256), 256, 0, st>>>(s, d, (long long)n4);
}
static inline void run_gemm(const unsigned short* A, int lda, const unsigned short* Bw,
                            int ldb, float* C, int ldc, int N, int K,
                            const float* bias, hipStream_t stream) {
  dim3 grid((N + GBN - 1) / GBN, MROWS / GBM);
  gemm_bf16_wmma<<<grid, 256, 0, stream>>>(A, lda, Bw, ldb, C, ldc, N, K, bias);
}

extern "C" void kernel_launch(void* const* d_in, const int* in_sizes, int n_in,
                              void* d_out, int out_size, void* d_ws, size_t ws_size,
                              hipStream_t stream) {
  (void)in_sizes; (void)n_in; (void)out_size; (void)ws_size;
  const float* x        = (const float*)d_in[0];
  const float* norm_w   = (const float*)d_in[1];
  const float* in_proj  = (const float*)d_in[2];
  const float* conv_w   = (const float*)d_in[3];
  const float* conv_b   = (const float*)d_in[4];
  const float* x_proj   = (const float*)d_in[5];
  const float* dt_head  = (const float*)d_in[6];
  const float* dt_b     = (const float*)d_in[7];
  const float* Dp       = (const float*)d_in[8];
  const float* reduce_w = (const float*)d_in[9];
  const float* q_net    = (const float*)d_in[10];
  const float* gain_q   = (const float*)d_in[11];
  const float* gain_k   = (const float*)d_in[12];
  const float* alpha    = (const float*)d_in[13];
  const float* beta     = (const float*)d_in[14];
  const float* qkv_w    = (const float*)d_in[15];
  const float* qkv_b    = (const float*)d_in[16];
  const float* lr_bias  = (const float*)d_in[17];
  const float* W1       = (const float*)d_in[18];
  const float* b1       = (const float*)d_in[19];
  const float* tok_idx  = (const float*)d_in[20];
  const float* tok_bias = (const float*)d_in[21];
  const float* ln_w     = (const float*)d_in[22];
  const float* ln_b     = (const float*)d_in[23];
  const float* pn_w     = (const float*)d_in[24];
  const float* pn_b     = (const float*)d_in[25];
  const float* wo_w     = (const float*)d_in[26];
  const float* wo_b     = (const float*)d_in[27];
  float* out = (float*)d_out;
  float* ws  = (float*)d_ws;

  // Workspace layout (float offsets), lifetime-based reuse:
  const size_t oXN   = 0;                    // xn -> later gq
  const size_t oXZ   = oXN + 4194304;        // xz -> later proj/ttt/gated/c_hid/z_ttt
  const size_t oXB   = oXZ + 50331648;       // xb (y in place) -> later z_long, gk
  const size_t oXDBL = oXB + 25165824;
  const size_t oDT   = oXDBL + 393216;
  const size_t oHRED = oDT + 25165824;
  const size_t oGQM  = oHRED + 4194304;
  const size_t oBFW  = oGQM + 2048;          // bf16 weights   (<= 3,145,728 elems)
  const size_t oBFA  = oBFW + 1572864;       // bf16 activations (<= 25,165,824 elems)
  const size_t oBFH  = oBFA + 12582912;      // bf16 h_red (long-lived)
  const size_t oBFX  = oBFH + 2097152;       // bf16 x_dbl
  float* xn    = ws + oXN;
  float* xz    = ws + oXZ;
  float* xb    = ws + oXB;
  float* xdbl  = ws + oXDBL;
  float* dt    = ws + oDT;
  float* hred  = ws + oHRED;
  float* gqm   = ws + oGQM;
  unsigned short* bfW = (unsigned short*)(ws + oBFW);
  unsigned short* bfA = (unsigned short*)(ws + oBFA);
  unsigned short* bfH = (unsigned short*)(ws + oBFH);
  unsigned short* bfX = (unsigned short*)(ws + oBFX);
  float* proj   = ws + oXZ;
  float* tttraw = ws + oXZ + 12648448;
  float* gated  = ws + oXZ + 16842752;
  float* chid   = ws + oXZ + 21037056;
  float* zttt   = ws + oXZ + 25231360;
  float* zlong  = ws + oXB;
  float* gk     = ws + oXB + 4194304;
  float* gq     = ws + oXN;

  // 1. RMSNorm
  rmsnorm_kernel<<<MROWS, 256, 0, stream>>>(x, norm_w, xn);
  // 2. in_proj (N=6144, K=512)
  cvt_bf16(xn, bfA, (size_t)MROWS * DM, stream);
  cvt_bf16(in_proj, bfW, (size_t)(2 * DI) * DM, stream);
  run_gemm(bfA, DM, bfW, DM, xz, 2 * DI, 2 * DI, DM, nullptr, stream);
  // 3. causal conv + SiLU
  {
    size_t total = (size_t)MROWS * DI;
    conv_silu_kernel<<<(unsigned)((total + 255) / 256), 256, 0, stream>>>(xz, conv_w, conv_b, xb);
  }
  // 4. x_proj (N=48, K=3072)
  cvt_bf16(xb, bfA, (size_t)MROWS * DI, stream);
  cvt_bf16(x_proj, bfW, (size_t)48 * DI, stream);
  run_gemm(bfA, DI, bfW, DI, xdbl, 48, 48, DI, nullptr, stream);
  // 5. dt (N=3072, K=32, A = x_dbl[:, :32] with lda=48)
  cvt_bf16(xdbl, bfX, (size_t)MROWS * 48, stream);
  cvt_bf16(dt_head, bfW, (size_t)DI * DT_RANK, stream);
  run_gemm(bfX, 48, bfW, DT_RANK, dt, DI, DI, DT_RANK, nullptr, stream);
  // 6. Longhorn scan (y overwrites xb)
  longhorn_scan_kernel<<<dim3(DI / 256, BB), 256, 0, stream>>>(xb, xz, dt, xdbl, dt_b, Dp);
  // 7. h_red (N=512, K=3072)
  cvt_bf16(xb, bfA, (size_t)MROWS * DI, stream);
  cvt_bf16(reduce_w, bfW, (size_t)DM * DI, stream);
  run_gemm(bfA, DI, bfW, DI, hred, DM, DM, DI, nullptr, stream);
  // 8. qkv proj (N=1544, K=512)    [xz region is dead now]
  cvt_bf16(hred, bfH, (size_t)MROWS * DM, stream);
  cvt_bf16(qkv_w, bfW, (size_t)1544 * DM, stream);
  run_gemm(bfH, DM, bfW, DM, proj, 1544, 1544, DM, qkv_b, stream);
  // 9. TTT scan
  ttt_scan_kernel<<<BB * NHEAD, HF, 0, stream>>>(proj, lr_bias, W1, b1, tok_idx,
                                                 tok_bias, ln_w, ln_b, tttraw);
  // 10. gate
  gate_kernel<<<MROWS, 256, 0, stream>>>(tttraw, proj, pn_w, pn_b, gated);
  // 11. c_hidden = gated @ wo^T + wo_b
  cvt_bf16(gated, bfA, (size_t)MROWS * DM, stream);
  cvt_bf16(wo_w, bfW, (size_t)DM * DM, stream);
  run_gemm(bfA, DM, bfW, DM, chid, DM, DM, DM, wo_b, stream);
  // 12. projections
  cvt_bf16(q_net, bfW, (size_t)DM * DM, stream);
  run_gemm(bfH, DM, bfW, DM, zlong, DM, DM, DM, nullptr, stream);
  cvt_bf16(chid, bfA, (size_t)MROWS * DM, stream);
  run_gemm(bfA, DM, bfW, DM, zttt, DM, DM, DM, nullptr, stream);
  cvt_bf16(gain_q, bfW, (size_t)DM * DM, stream);
  run_gemm(bfH, DM, bfW, DM, gq, DM, DM, DM, nullptr, stream);
  cvt_bf16(gain_k, bfW, (size_t)DM * DM, stream);
  run_gemm(bfH, DM, bfW, DM, gk, DM, DM, DM, nullptr, stream);
  // 13-14. gain + outputs
  colmean_kernel<<<BB, DM, 0, stream>>>(gq, gqm);
  out_kernel<<<MROWS, 256, 0, stream>>>(zlong, zttt, gk, gqm, x, alpha, beta, out);
}
